// SpatialBlock_89309549953892
// MI455X (gfx1250) — compile-verified, matrix operand order checked by reading, not run on hardware
//
#include <hip/hip_runtime.h>
#include <hip/hip_bf16.h>
#include <math.h>
#include <stdint.h>

typedef __attribute__((ext_vector_type(16))) __bf16 v16bf;
typedef __attribute__((ext_vector_type(8)))  __bf16 v8bf;
typedef __attribute__((ext_vector_type(8)))  float  v8f;

static constexpr int  B_   = 8;
static constexpr int  C_   = 512;
static constexpr int  N_   = 8190;
static constexpr int  NP_  = 8192;     // padded N (WS=4)
static constexpr int  H_   = 16;
static constexpr int  HID_ = 2048;
static constexpr long long MTOT = (long long)B_ * NP_;   // 65536 rows

static constexpr int KSTAGE  = 512;    // K chunk staged in LDS per fill
static constexpr int BSTRIDE = 520;    // padded LDS row stride (bank-conflict-free b128)

// ---------------------------------------------------------------- utilities
__global__ __launch_bounds__(256) void k_f32_to_bf16(const float* __restrict__ s,
                                                     __bf16* __restrict__ d, int n) {
  int i = blockIdx.x * 256 + threadIdx.x;
  if (i < n) d[i] = (__bf16)s[i];
}

// ------------------------------------------------ cpe1: x + dwconv(x), NCH -> (B,NP,C)
__global__ __launch_bounds__(256) void k_cpe1(const float* __restrict__ x,
                                              const float* __restrict__ w,
                                              const float* __restrict__ bias,
                                              float* __restrict__ outT) {
  __shared__ float t[64][65];
  const int n0 = blockIdx.x * 64;
  const int c0 = blockIdx.y * 64;
  const int b  = blockIdx.z;
  const int tn = threadIdx.x & 63;
  const int tg = threadIdx.x >> 6;     // 0..3
  const int n  = n0 + tn;
  for (int cc = tg; cc < 64; cc += 4) {
    const int c = c0 + cc;
    float v = 0.f;
    if (n < N_) {
      const float* row = x + ((long long)b * C_ + c) * N_;
      float xm = (n > 0)      ? row[n - 1] : 0.f;
      float x0 = row[n];
      float xp = (n + 1 < N_) ? row[n + 1] : 0.f;
      v = x0 + w[c * 3 + 0] * xm + w[c * 3 + 1] * x0 + w[c * 3 + 2] * xp + bias[c];
    }
    t[cc][tn] = v;
  }
  __syncthreads();
  for (int nn = tg; nn < 64; nn += 4) {
    outT[((long long)b * NP_ + n0 + nn) * C_ + c0 + tn] = t[tn][nn];
  }
}

// ------------------------------------------------ LayerNorm (f32 in -> bf16 out), pad rows -> 0
__global__ __launch_bounds__(256) void k_ln(const float* __restrict__ in,
                                            const float* __restrict__ g,
                                            const float* __restrict__ bt,
                                            __bf16* __restrict__ out) {
  const int row  = blockIdx.x * 8 + (threadIdx.x >> 5);
  const int lane = threadIdx.x & 31;
  const float* p = in + (long long)row * C_;
  float xv[16];
  float s = 0.f, ss = 0.f;
  for (int j = 0; j < 16; ++j) {
    float x = p[j * 32 + lane];
    xv[j] = x; s += x; ss += x * x;
  }
  for (int off = 16; off; off >>= 1) {
    s  += __shfl_xor(s,  off, 32);
    ss += __shfl_xor(ss, off, 32);
  }
  const float mean = s * (1.f / C_);
  const float var  = ss * (1.f / C_) - mean * mean;
  const float inv  = rsqrtf(var + 1e-5f);
  __bf16* q = out + (long long)row * C_;
  const int tok = row & (NP_ - 1);
  if (tok < N_) {
    for (int j = 0; j < 16; ++j) {
      const int c = j * 32 + lane;
      q[c] = (__bf16)((xv[j] - mean) * inv * g[c] + bt[c]);
    }
  } else {
    for (int j = 0; j < 16; ++j) q[j * 32 + lane] = (__bf16)0.f;
  }
}

// ------------------------------------------------ windowed attention, one wave per (b, window, head)
__global__ __launch_bounds__(256) void k_attn(const __bf16* __restrict__ qkv,
                                              __bf16* __restrict__ o) {
  const int gw   = blockIdx.x * 8 + (threadIdx.x >> 5);
  const int lane = threadIdx.x & 31;
  const int h = gw & 15;
  const int w = (gw >> 4) & 2047;
  const int b = gw >> 15;
  const long long tok0 = (long long)b * NP_ + (long long)w * 4;
  const __bf16* base = qkv + tok0 * (3 * C_) + h * 32 + lane;
  float q[4], k[4], v[4];
  for (int i = 0; i < 4; ++i) {
    q[i] = (float)base[(long long)i * (3 * C_) + 0 * C_];
    k[i] = (float)base[(long long)i * (3 * C_) + 1 * C_];
    v[i] = (float)base[(long long)i * (3 * C_) + 2 * C_];
  }
  float sc[16];
  for (int i = 0; i < 4; ++i)
    for (int j = 0; j < 4; ++j) sc[i * 4 + j] = q[i] * k[j];
  for (int off = 16; off; off >>= 1)
    for (int t = 0; t < 16; ++t) sc[t] += __shfl_xor(sc[t], off, 32);
  const float scale = 0.17677669529663687f;   // 1/sqrt(32)
  __bf16* ob = o + tok0 * C_ + h * 32 + lane;
  for (int i = 0; i < 4; ++i) {
    float p0 = sc[i * 4 + 0] * scale, p1 = sc[i * 4 + 1] * scale;
    float p2 = sc[i * 4 + 2] * scale, p3 = sc[i * 4 + 3] * scale;
    float m = fmaxf(fmaxf(p0, p1), fmaxf(p2, p3));
    float e0 = expf(p0 - m), e1 = expf(p1 - m), e2 = expf(p2 - m), e3 = expf(p3 - m);
    float rs = 1.f / (e0 + e1 + e2 + e3);
    float ov = (e0 * v[0] + e1 * v[1] + e2 * v[2] + e3 * v[3]) * rs;
    ob[(long long)i * C_] = (__bf16)ov;
  }
}

// ------------------------------------------------ cpe2: xt = x1 + dwconv(x1) in (B,NP,C)
__global__ __launch_bounds__(256) void k_cpe2(const float* __restrict__ x1,
                                              const float* __restrict__ w,
                                              const float* __restrict__ bias,
                                              float* __restrict__ xt) {
  const int rid = blockIdx.x;
  const int n   = rid & (NP_ - 1);
  for (int c = threadIdx.x; c < C_; c += 256) {
    float v = 0.f;
    if (n < N_) {
      const long long base = (long long)rid * C_ + c;
      float x0 = x1[base];
      float xm = (n > 0) ? x1[base - C_] : 0.f;
      float xp = x1[base + C_];            // row N is zeroed pad -> correct boundary
      v = x0 + w[c * 3 + 0] * xm + w[c * 3 + 1] * x0 + w[c * 3 + 2] * xp + bias[c];
    }
    xt[(long long)rid * C_ + c] = v;
  }
}

// ------------------------------------------------ WMMA fragment loaders (16-bit layout swizzle)
__device__ inline v16bf load_fragA(const __bf16* __restrict__ p, long long row0,
                                   int ld, int k0, int lane) {
  const long long r  = row0 + (lane & 15);
  const int       kb = k0 + ((lane >> 4) << 3);
  const __bf16*   q  = p + r * (long long)ld + kb;
  v8bf lo = *(const v8bf*)(q);
  v8bf hi = *(const v8bf*)(q + 16);
  return __builtin_shufflevector(lo, hi, 0, 1, 2, 3, 4, 5, 6, 7,
                                 8, 9, 10, 11, 12, 13, 14, 15);
}

__device__ inline v16bf load_fragB_lds(const __bf16* __restrict__ Bs, int nrow0,
                                       int kk, int lane) {
  const int r  = nrow0 + (lane & 15);
  const int kb = kk + ((lane >> 4) << 3);
  const __bf16* q = Bs + r * BSTRIDE + kb;
  v8bf lo = *(const v8bf*)(q);
  v8bf hi = *(const v8bf*)(q + 16);
  return __builtin_shufflevector(lo, hi, 0, 1, 2, 3, 4, 5, 6, 7,
                                 8, 9, 10, 11, 12, 13, 14, 15);
}

// EPI: 0 = bf16 out (+bias)         [qkv]
//      1 = f32 out = res + v, pad rows zeroed  [proj -> x1 in-place]
//      2 = bf16 out = GELU(v+bias)  [fc1]
//      3 = final: d_out[b][c][n] = res + v + bias (real tokens only)  [fc2]
template <int EPI>
__device__ inline void epilogue(const v8f& acc, long long m0, int n0, int lane, int Nt,
                                const float* __restrict__ bias,
                                float* __restrict__ fout, __bf16* __restrict__ bfout,
                                const float* __restrict__ res) {
  const int   ncol = n0 + (lane & 15);
  const float bv   = bias ? bias[ncol] : 0.f;
  const int   mo   = (lane >> 4) * 8;
  for (int r = 0; r < 8; ++r) {
    const long long m = m0 + r + mo;
    float v = acc[r] + bv;
    if (EPI == 0) {
      bfout[m * Nt + ncol] = (__bf16)v;
    } else if (EPI == 1) {
      const int tok = (int)(m & (NP_ - 1));
      fout[m * Nt + ncol] = (tok < N_) ? (res[m * Nt + ncol] + v) : 0.f;
    } else if (EPI == 2) {
      float g = 0.5f * v * (1.f + erff(v * 0.70710678118654752f));
      bfout[m * Nt + ncol] = (__bf16)g;
    } else {
      const int tok = (int)(m & (NP_ - 1));
      if (tok < N_) {
        const long long bb = m >> 13;
        fout[(bb * Nt + ncol) * (long long)N_ + tok] = res[m * Nt + ncol] + v;
      }
    }
  }
}

// ------------------------------------------------ tiled WMMA GEMM: D = A(MxK) * W(NtxK)^T (+epi)
// Weight panel (64 rows x KSTAGE) staged into LDS via GLOBAL_LOAD_ASYNC_TO_LDS_B128
// (ASYNCcnt-tracked, no VGPR roundtrip); A fragments streamed from global + prefetch.
template <int EPI>
__global__ __launch_bounds__(256) void gemm_wmma(const __bf16* __restrict__ A,
                                                 const __bf16* __restrict__ W,
                                                 const float* __restrict__ bias,
                                                 int K, int Nt,
                                                 float* __restrict__ fout,
                                                 __bf16* __restrict__ bfout,
                                                 const float* __restrict__ res) {
  __shared__ __bf16 Bs[64 * BSTRIDE];                 // ~65 KB, padded rows

  const int tid  = threadIdx.x;
  const int lane = tid & 31;
  const int wid  = tid >> 5;               // 8 waves
  const int wm   = wid & 3;                // 4 wave rows (M)
  const int wn   = wid >> 2;               // 2 wave cols (N)
  const long long m0    = (long long)blockIdx.x * 128 + wm * 32;
  const int       n0blk = blockIdx.y * 64;
  const int       n0w   = wn * 32;         // within-block N offset for this wave

  // async-copy mapping: each thread owns a 256B quarter-row of the weight panel
  const int crow = tid >> 2;               // 0..63
  const int cq   = tid & 3;                // 0..3

  v8f acc00 = {}, acc01 = {}, acc10 = {}, acc11 = {};

  for (int ks = 0; ks < K; ks += KSTAGE) {
    // ---- fill LDS weight panel with async vector copies (ASYNCcnt path)
    {
      unsigned lds = (unsigned)(uintptr_t)(&Bs[crow * BSTRIDE + cq * 128]);
      unsigned long long g =
          (unsigned long long)(uintptr_t)(W + ((long long)(n0blk + crow)) * K + ks + cq * 128);
      #pragma unroll
      for (int i = 0; i < 16; ++i) {       // 16 x b128 per lane = 256B
        asm volatile("global_load_async_to_lds_b128 %0, %1, off"
                     :: "v"(lds), "v"(g) : "memory");
        lds += 16; g += 16;
      }
      asm volatile("s_wait_asynccnt 0x0" ::: "memory");
    }
    __syncthreads();

    // ---- consume: K inner loop over the staged panel
    for (int kk = 0; kk < KSTAGE; kk += 32) {
      if (kk + 32 < KSTAGE) {              // prefetch next A panel (global_prefetch path)
        __builtin_prefetch((const void*)(A + (m0 + (lane & 15)) * (long long)K + ks + kk + 32), 0, 1);
      }
      v16bf a0 = load_fragA(A, m0,      K, ks + kk, lane);
      v16bf a1 = load_fragA(A, m0 + 16, K, ks + kk, lane);
      v16bf b0 = load_fragB_lds(Bs, n0w,      kk, lane);
      v16bf b1 = load_fragB_lds(Bs, n0w + 16, kk, lane);
      acc00 = __builtin_amdgcn_wmma_f32_16x16x32_bf16(false, a0, false, b0, (short)0, acc00, false, false);
      acc01 = __builtin_amdgcn_wmma_f32_16x16x32_bf16(false, a0, false, b1, (short)0, acc01, false, false);
      acc10 = __builtin_amdgcn_wmma_f32_16x16x32_bf16(false, a1, false, b0, (short)0, acc10, false, false);
      acc11 = __builtin_amdgcn_wmma_f32_16x16x32_bf16(false, a1, false, b1, (short)0, acc11, false, false);
    }
    __syncthreads();                       // panel consumed before next fill
  }

  const int n0 = n0blk + n0w;
  epilogue<EPI>(acc00, m0,      n0,      lane, Nt, bias, fout, bfout, res);
  epilogue<EPI>(acc01, m0,      n0 + 16, lane, Nt, bias, fout, bfout, res);
  epilogue<EPI>(acc10, m0 + 16, n0,      lane, Nt, bias, fout, bfout, res);
  epilogue<EPI>(acc11, m0 + 16, n0 + 16, lane, Nt, bias, fout, bfout, res);
}

// ---------------------------------------------------------------- launcher
extern "C" void kernel_launch(void* const* d_in, const int* in_sizes, int n_in,
                              void* d_out, int out_size, void* d_ws, size_t ws_size,
                              hipStream_t stream) {
  const float* x      = (const float*)d_in[0];
  const float* cpe1_w = (const float*)d_in[1];
  const float* cpe1_b = (const float*)d_in[2];
  const float* ln1_g  = (const float*)d_in[3];
  const float* ln1_b  = (const float*)d_in[4];
  const float* qkv_w  = (const float*)d_in[5];
  const float* qkv_b  = (const float*)d_in[6];
  const float* proj_w = (const float*)d_in[7];
  const float* proj_b = (const float*)d_in[8];
  const float* cpe2_w = (const float*)d_in[9];
  const float* cpe2_b = (const float*)d_in[10];
  const float* ln2_g  = (const float*)d_in[11];
  const float* ln2_b  = (const float*)d_in[12];
  const float* fc1_w  = (const float*)d_in[13];
  const float* fc1_b  = (const float*)d_in[14];
  const float* fc2_w  = (const float*)d_in[15];
  const float* fc2_b  = (const float*)d_in[16];
  float* out = (float*)d_out;

  char* ws = (char*)d_ws;
  const size_t SZ_F32 = (size_t)MTOT * C_ * 4;           // 134 MB
  const size_t SZ_BF  = (size_t)MTOT * C_ * 2;           // 67 MB
  float*  buf1 = (float*)(ws);                           // shortcut -> x1 (in-place)
  float*  buf2 = (float*)(ws + SZ_F32);                  // xt
  __bf16* hbuf = (__bf16*)(ws + 2 * SZ_F32);             // h / hm (bf16)
  __bf16* qbuf = (__bf16*)(ws + 2 * SZ_F32 + SZ_BF);     // qkv / fc1 act (268 MB)
  __bf16* obuf = (__bf16*)(ws + 2 * SZ_F32 + SZ_BF + (size_t)MTOT * HID_ * 2);
  char*   wb   = (char*)(obuf) + SZ_BF;
  __bf16* qkv_wb  = (__bf16*)(wb);
  __bf16* proj_wb = (__bf16*)(wb + (size_t)3 * C_ * C_ * 2);
  __bf16* fc1_wb  = (__bf16*)(wb + (size_t)3 * C_ * C_ * 2 + (size_t)C_ * C_ * 2);
  __bf16* fc2_wb  = (__bf16*)(wb + (size_t)3 * C_ * C_ * 2 + (size_t)C_ * C_ * 2 +
                              (size_t)HID_ * C_ * 2);

  // 0) weights -> bf16
  k_f32_to_bf16<<<(3 * C_ * C_ + 255) / 256, 256, 0, stream>>>(qkv_w, qkv_wb, 3 * C_ * C_);
  k_f32_to_bf16<<<(C_ * C_ + 255) / 256, 256, 0, stream>>>(proj_w, proj_wb, C_ * C_);
  k_f32_to_bf16<<<(HID_ * C_ + 255) / 256, 256, 0, stream>>>(fc1_w, fc1_wb, HID_ * C_);
  k_f32_to_bf16<<<(C_ * HID_ + 255) / 256, 256, 0, stream>>>(fc2_w, fc2_wb, C_ * HID_);

  // 1) shortcut = transpose(x + dwconv1(x))  (pad rows zeroed)
  k_cpe1<<<dim3(NP_ / 64, C_ / 64, B_), 256, 0, stream>>>(x, cpe1_w, cpe1_b, buf1);

  // 2) h = LN1(shortcut) -> bf16
  k_ln<<<(int)(MTOT / 8), 256, 0, stream>>>(buf1, ln1_g, ln1_b, hbuf);

  // 3) qkv = h @ qkv_w^T + b   (M=65536, N=1536, K=512)
  gemm_wmma<0><<<dim3((int)(MTOT / 128), (3 * C_) / 64), 256, 0, stream>>>(
      hbuf, qkv_wb, qkv_b, C_, 3 * C_, nullptr, qbuf, nullptr);

  // 4) windowed attention -> o (bf16)
  k_attn<<<(int)(B_ * 2048 * H_ / 8), 256, 0, stream>>>(qbuf, obuf);

  // 5) x1 = shortcut + o @ proj_w^T + b   (in-place on buf1, pads zeroed)
  gemm_wmma<1><<<dim3((int)(MTOT / 128), C_ / 64), 256, 0, stream>>>(
      obuf, proj_wb, proj_b, C_, C_, buf1, nullptr, buf1);

  // 6) xt = x1 + dwconv2(x1)
  k_cpe2<<<(int)MTOT, 256, 0, stream>>>(buf1, cpe2_w, cpe2_b, buf2);

  // 7) hm = LN2(xt) -> bf16
  k_ln<<<(int)(MTOT / 8), 256, 0, stream>>>(buf2, ln2_g, ln2_b, hbuf);

  // 8) act = GELU(hm @ fc1_w^T + b)  (M=65536, N=2048, K=512)
  gemm_wmma<2><<<dim3((int)(MTOT / 128), HID_ / 64), 256, 0, stream>>>(
      hbuf, fc1_wb, fc1_b, C_, HID_, nullptr, qbuf, nullptr);

  // 9) out[b][c][n] = xt + act @ fc2_w^T + b  (K=2048, 4 LDS stages)
  gemm_wmma<3><<<dim3((int)(MTOT / 128), C_ / 64), 256, 0, stream>>>(
      qbuf, fc2_wb, fc2_b, HID_, C_, out, nullptr, buf2);

  (void)in_sizes; (void)n_in; (void)out_size; (void)ws_size;
}
